// TransformerAttentionModule_58342835749559
// MI455X (gfx1250) — compile-verified, matrix operand order checked by reading, not run on hardware
//
#include <hip/hip_runtime.h>

// ---------------------------------------------------------------------------
// Graph multi-head attention for MI455X (gfx1250, wave32).
//   1) pack x, Wqkv -> f16 in WMMA per-lane layout (one-time, coalesced)
//   2) qkv = x @ Wqkv + bqkv       -> v_wmma_f32_16x16x32_f16, 5 M-tiles/wave
//   3) edge scores + segment softmax -> float4 gathers + hw atomics
//   4) agg scatter-add (v * prob)    -> GLOBAL_ATOMIC_ADD_F32
//   5) pack agg, Wout; out = agg @ Wout + bout -> WMMA
// All GEMM dims exact multiples (100000=16*6250=80*1250, K=128, N=384/128),
// so EXEC stays all-ones (WMMA requirement) and there are no tails.
// ---------------------------------------------------------------------------

typedef __attribute__((ext_vector_type(16))) _Float16 v16h;
typedef __attribute__((ext_vector_type(8)))  float    v8f;

#define NN      100000
#define NE      1600000
#define DIMM    128
#define NHEAD   8
#define HDIM    16
#define QKVCOLS 384           // 8 heads * (q16|k16|v16), matches reference reshape
#define KB32    (DIMM / 32)   // 4 k-steps of 32
#define MPW     5             // M-tiles per wave (6250 / 5 = 1250, exact)

// Monotonic order-preserving float<->uint mapping for atomic segment-max.
__device__ __forceinline__ unsigned flipF(float f) {
  unsigned u = __float_as_uint(f);
  return (u & 0x80000000u) ? ~u : (u | 0x80000000u);
}
__device__ __forceinline__ float unflipF(unsigned u) {
  return (u & 0x80000000u) ? __uint_as_float(u & 0x7FFFFFFFu)
                           : __uint_as_float(~u);
}

// ---------------------------------------------------------------------------
// Pack fp32 matrices into the exact CDNA5 WMMA 16-bit per-lane layout
// (ISA 7.12.2): element e of lane L covers K = kb*32 + (L>>4)*8 + (e<8?e:e+8).
// Flat packed index: ((tile*KB + kb)*32 + lane)*16 + e  -> coalesced writes.
// ---------------------------------------------------------------------------
// A-side: tile indexes 16 rows; source element A[row, k], row-major A[M,K].
__global__ void pack_a_f16(const float* __restrict__ A, _Float16* __restrict__ Ap,
                           int K, int total) {
  int idx = blockIdx.x * blockDim.x + threadIdx.x;
  if (idx >= total) return;
  int e    = idx & 15;
  int lane = (idx >> 4) & 31;
  int rest = idx >> 9;
  int kb   = rest & (KB32 - 1);          // KB32 = 4, power of two
  int mt   = rest >> 2;
  int row  = mt * 16 + (lane & 15);
  int k    = kb * 32 + (lane >> 4) * 8 + ((e < 8) ? e : e + 8);
  Ap[idx] = (_Float16)A[(size_t)row * K + k];
}

// B-side: tile indexes 16 cols; source element B[k, col], row-major B[K,N].
__global__ void pack_b_f16(const float* __restrict__ B, _Float16* __restrict__ Bp,
                           int N, int total) {
  int idx = blockIdx.x * blockDim.x + threadIdx.x;
  if (idx >= total) return;
  int e    = idx & 15;
  int lane = (idx >> 4) & 31;
  int rest = idx >> 9;
  int kb   = rest & (KB32 - 1);
  int nt   = rest >> 2;
  int col  = nt * 16 + (lane & 15);
  int k    = kb * 32 + (lane >> 4) * 8 + ((e < 8) ? e : e + 8);
  Bp[idx] = (_Float16)B[(size_t)k * N + col];
}

// ---------------------------------------------------------------------------
// C[M,N] = A@B + bias from pre-packed f16 operands. One wave covers MPW
// consecutive 16-row tiles at one 16-col tile: B is fetched once per k-step
// (2x global_load_b128, lane-contiguous 1KB/wave) and reused by MPW WMMAs.
// 4 waves/block -> 4 N-tiles. K-loop fully unrolled: 20 static v_wmma.
// ---------------------------------------------------------------------------
__global__ void __launch_bounds__(128)
gemm_packed_wmma(const v16h* __restrict__ Ap, const v16h* __restrict__ Bp,
                 const float* __restrict__ bias, float* __restrict__ C, int N) {
  const int lane   = threadIdx.x & 31;
  const int wave   = threadIdx.x >> 5;
  const int half   = lane >> 4;
  const int sub    = lane & 15;
  const int tile_n = blockIdx.y * 4 + wave;
  const int mt0    = blockIdx.x * MPW;

  v8f acc[MPW] = {};
#pragma unroll
  for (int kb = 0; kb < KB32; ++kb) {
    const v16h b = Bp[((size_t)tile_n * KB32 + kb) * 32 + lane];
#pragma unroll
    for (int m = 0; m < MPW; ++m) {
      const v16h a = Ap[((size_t)(mt0 + m) * KB32 + kb) * 32 + lane];
      // (neg_a, A, neg_b, B, c_mod, C, reuse_a, reuse_b)
      acc[m] = __builtin_amdgcn_wmma_f32_16x16x32_f16(false, a, false, b,
                                                      (short)0, acc[m],
                                                      false, false);
    }
  }

  const int   col = tile_n * 16 + sub;
  const float bv  = bias[col];
#pragma unroll
  for (int m = 0; m < MPW; ++m) {
#pragma unroll
    for (int r = 0; r < 8; ++r) {   // C VGPR r -> row r (lanes 0-15) / r+8
      C[(size_t)((mt0 + m) * 16 + half * 8 + r) * N + col] = acc[m][r] + bv;
    }
  }
}

// ---------------------------------------------------------------------------
// Softmax state init (workspace is poisoned by the harness; re-init per call).
// ---------------------------------------------------------------------------
__global__ void init_state(unsigned* __restrict__ mbuf,
                           float* __restrict__ denom,
                           float* __restrict__ agg) {
  int i = blockIdx.x * blockDim.x + threadIdx.x;
  if (i < NN * NHEAD) { mbuf[i] = 0u; denom[i] = 0.0f; }  // 0u < flip(any real)
  if (i < NN * DIMM)  agg[i] = 0.0f;
}

// ---------------------------------------------------------------------------
// score[e,h] = dot(k[src], q[dst]) / sqrt(16); atomic segment-max over dst.
// One thread per (edge, head): 2 x 64B contiguous gathers (float4 x4).
// ---------------------------------------------------------------------------
__global__ void edge_scores(const float* __restrict__ qkv,
                            const int* __restrict__ src,
                            const int* __restrict__ dst,
                            float* __restrict__ scores,
                            unsigned* __restrict__ mbuf) {
  int t = blockIdx.x * blockDim.x + threadIdx.x;
  if (t >= NE * NHEAD) return;
  int e = t >> 3, h = t & 7;
  int s = src[e], d = dst[e];
  const float4* kk = (const float4*)(qkv + (size_t)s * QKVCOLS + h * 48 + 16);
  const float4* qq = (const float4*)(qkv + (size_t)d * QKVCOLS + h * 48);
  float acc = 0.0f;
#pragma unroll
  for (int i = 0; i < 4; ++i) {
    float4 kv = kk[i], qv = qq[i];
    acc += kv.x * qv.x + kv.y * qv.y + kv.z * qv.z + kv.w * qv.w;
  }
  float sc = acc * 0.25f;  // 1/sqrt(HDIM)
  scores[t] = sc;
  atomicMax(&mbuf[d * NHEAD + h], flipF(sc));
}

// p = exp(score - m[dst]); denom[dst] += p  (scores overwritten with p)
__global__ void edge_expsum(const int* __restrict__ dst,
                            float* __restrict__ scores,
                            const unsigned* __restrict__ mbuf,
                            float* __restrict__ denom) {
  int t = blockIdx.x * blockDim.x + threadIdx.x;
  if (t >= NE * NHEAD) return;
  int e = t >> 3, h = t & 7;
  int d = dst[e];
  float p = expf(scores[t] - unflipF(mbuf[d * NHEAD + h]));
  scores[t] = p;
  atomicAdd(&denom[d * NHEAD + h], p);
}

// agg[dst, h*16..] += v[src, h] * (p / denom[dst])
__global__ void edge_aggregate(const float* __restrict__ qkv,
                               const int* __restrict__ src,
                               const int* __restrict__ dst,
                               const float* __restrict__ scores,
                               const float* __restrict__ denom,
                               float* __restrict__ agg) {
  int t = blockIdx.x * blockDim.x + threadIdx.x;
  if (t >= NE * NHEAD) return;
  int e = t >> 3, h = t & 7;
  int s = src[e], d = dst[e];
  float prob = scores[t] / denom[d * NHEAD + h];
  const float4* vv = (const float4*)(qkv + (size_t)s * QKVCOLS + h * 48 + 32);
  float* ab = agg + (size_t)d * DIMM + h * HDIM;
#pragma unroll
  for (int i = 0; i < 4; ++i) {
    float4 v4 = vv[i];
    atomicAdd(ab + i * 4 + 0, v4.x * prob);
    atomicAdd(ab + i * 4 + 1, v4.y * prob);
    atomicAdd(ab + i * 4 + 2, v4.z * prob);
    atomicAdd(ab + i * 4 + 3, v4.w * prob);
  }
}

// ---------------------------------------------------------------------------
extern "C" void kernel_launch(void* const* d_in, const int* in_sizes, int n_in,
                              void* d_out, int out_size, void* d_ws, size_t ws_size,
                              hipStream_t stream) {
  const float* x    = (const float*)d_in[0];
  const float* Wqkv = (const float*)d_in[1];
  const float* bqkv = (const float*)d_in[2];
  const float* Wout = (const float*)d_in[3];
  const float* bout = (const float*)d_in[4];
  const int*   src  = (const int*)d_in[5];
  const int*   dst  = (const int*)d_in[6];
  float* out = (float*)d_out;

  // ---- workspace layout -------------------------------------------------
  const size_t MTILES = NN / 16;                       // 6250
  const size_t apackE = MTILES * KB32 * 32 * 16;       // 12.8M halves (shared)
  const size_t bpQkvE = (QKVCOLS / 16) * KB32 * 32 * 16; // 49152 halves
  const size_t bpOutE = (DIMM / 16) * KB32 * 32 * 16;    // 16384 halves

  float*    qkv    = (float*)d_ws;                       // NN*384 f32
  float*    scores = qkv + (size_t)NN * QKVCOLS;         // NE*8   f32
  unsigned* mbuf   = (unsigned*)(scores + (size_t)NE * NHEAD);
  float*    denom  = (float*)(mbuf + (size_t)NN * NHEAD);
  float*    agg    = denom + (size_t)NN * NHEAD;         // NN*128 f32
  _Float16* Apack  = (_Float16*)(agg + (size_t)NN * DIMM); // reused by both GEMMs
  _Float16* BpQkv  = Apack + apackE;
  _Float16* BpOut  = BpQkv + bpQkvE;

  const int ET = NE * NHEAD;                           // 12.8M (edge,head)

  // ---- 1) pack weights + x, run QKV GEMM --------------------------------
  pack_b_f16<<<(int)((bpQkvE + 255) / 256), 256, 0, stream>>>(Wqkv, BpQkv,
                                                              QKVCOLS, (int)bpQkvE);
  pack_b_f16<<<(int)((bpOutE + 255) / 256), 256, 0, stream>>>(Wout, BpOut,
                                                              DIMM, (int)bpOutE);
  pack_a_f16<<<(int)((apackE + 255) / 256), 256, 0, stream>>>(x, Apack,
                                                              DIMM, (int)apackE);
  gemm_packed_wmma<<<dim3(MTILES / MPW, QKVCOLS / 16 / 4), 128, 0, stream>>>(
      (const v16h*)Apack, (const v16h*)BpQkv, bqkv, qkv, QKVCOLS);

  // ---- 2) edge softmax + aggregation ------------------------------------
  init_state<<<(NN * DIMM + 255) / 256, 256, 0, stream>>>(mbuf, denom, agg);
  edge_scores<<<(ET + 255) / 256, 256, 0, stream>>>(qkv, src, dst, scores, mbuf);
  edge_expsum<<<(ET + 255) / 256, 256, 0, stream>>>(dst, scores, mbuf, denom);
  edge_aggregate<<<(ET + 255) / 256, 256, 0, stream>>>(qkv, src, dst, scores,
                                                       denom, agg);

  // ---- 3) output projection ---------------------------------------------
  pack_a_f16<<<(int)((apackE + 255) / 256), 256, 0, stream>>>(agg, Apack,
                                                              DIMM, (int)apackE);
  gemm_packed_wmma<<<dim3(MTILES / MPW, DIMM / 16 / 4), 128, 0, stream>>>(
      (const v16h*)Apack, (const v16h*)BpOut, bout, out, DIMM);
}
